// LinearRNN_15857019257160
// MI455X (gfx1250) — compile-verified, matrix operand order checked by reading, not run on hardware
//
#include <hip/hip_runtime.h>
#include <hip/hip_bf16.h>
#include <cstdint>
#include <cstddef>

// ---------------- problem constants ----------------
constexpr int T_  = 32768;
constexpr int IN_ = 512;
constexpr int H_  = 1024;
constexpr int OUT_= 512;
constexpr int L_  = 4;

// scan blocking
constexpr int SR  = 128;          // rows per scan chunk
constexpr int SC  = T_ / SR;      // 256 chunks

// gemm blocking
constexpr int BM = 128, BN = 64, BK = 32;
constexpr int LDS_STRIDE = BK + 8;   // 40 bf16 = 80B rows, 16B aligned, conflict-friendly

typedef __bf16 bf16;
typedef __attribute__((ext_vector_type(16))) __bf16 v16bf;
typedef __attribute__((ext_vector_type(8)))  __bf16 v8bf;
typedef __attribute__((ext_vector_type(4)))  __bf16 v4bf;
typedef __attribute__((ext_vector_type(8)))  float  v8f;
typedef __attribute__((ext_vector_type(4)))  float  v4f;

// ---------------- helpers ----------------
__device__ __forceinline__ bf16 f2bf(float f) {
    union { float f; uint32_t u; } a; a.f = f;
    uint32_t r = (a.u + 0x7FFFu + ((a.u >> 16) & 1u)) >> 16;   // RNE
    union { uint16_t s; bf16 b; } o; o.s = (uint16_t)r;
    return o.b;
}

__device__ __forceinline__ v16bf mk16(v8bf lo, v8bf hi) {
    v16bf r;
#pragma unroll
    for (int e = 0; e < 8; ++e) { r[e] = lo[e]; r[e + 8] = hi[e]; }
    return r;
}

__device__ __forceinline__ void storeC(float* p, float v) { *p = v; }
__device__ __forceinline__ void storeC(bf16*  p, float v) { *p = f2bf(v); }

// ---------------- f32 -> bf16 convert ----------------
__global__ void k_f32_to_bf16(const float* __restrict__ src, bf16* __restrict__ dst, int n) {
    for (int i = blockIdx.x * blockDim.x + threadIdx.x; i < n; i += gridDim.x * blockDim.x)
        dst[i] = f2bf(src[i]);
}

// ---------------- segmented scan (3-pass blocked) ----------------
// pass 1: per-chunk segmented sum aggregate + "chunk has a reset flag"
__global__ void k_scan_chunk_agg(const bf16* __restrict__ in, const uint8_t* __restrict__ start,
                                 float* __restrict__ agg, int* __restrict__ aggflag) {
    __shared__ uint8_t fl[SR];
    const int chunk = blockIdx.x, tid = threadIdx.x;
    if (tid < SR) fl[tid] = start[chunk * SR + tid];
    __syncthreads();
    const int ch0 = tid * 4;
    float r0 = 0.f, r1 = 0.f, r2 = 0.f, r3 = 0.f;
    for (int r = 0; r < SR; ++r) {
        if (fl[r]) { r0 = r1 = r2 = r3 = 0.f; }
        v4bf v = *(const v4bf*)(in + (size_t)(chunk * SR + r) * H_ + ch0);
        r0 += (float)v[0]; r1 += (float)v[1]; r2 += (float)v[2]; r3 += (float)v[3];
    }
    float* ap = agg + (size_t)chunk * H_ + ch0;
    ap[0] = r0; ap[1] = r1; ap[2] = r2; ap[3] = r3;
    if (tid == 0) {
        int any = 0;
        for (int r = 0; r < SR; ++r) any |= fl[r];
        aggflag[chunk] = any;
    }
}

// pass 2: serial scan across chunk aggregates -> carry-in per chunk
__global__ void k_scan_carries(const float* __restrict__ agg, const int* __restrict__ aggflag,
                               const float* __restrict__ h0, float* __restrict__ carry) {
    const int ch0 = threadIdx.x * 4;
    float c0 = h0[ch0 + 0], c1 = h0[ch0 + 1], c2 = h0[ch0 + 2], c3 = h0[ch0 + 3];
    for (int c = 0; c < SC; ++c) {
        float* cp = carry + (size_t)c * H_ + ch0;
        cp[0] = c0; cp[1] = c1; cp[2] = c2; cp[3] = c3;
        const float* ap = agg + (size_t)c * H_ + ch0;
        if (aggflag[c]) { c0 = ap[0]; c1 = ap[1]; c2 = ap[2]; c3 = ap[3]; }
        else            { c0 += ap[0]; c1 += ap[1]; c2 += ap[2]; c3 += ap[3]; }
    }
}

// pass 3: redo local segmented scan seeded with carry, emit bf16 z
__global__ void k_scan_apply(const bf16* __restrict__ in, const uint8_t* __restrict__ start,
                             const float* __restrict__ carry, bf16* __restrict__ z) {
    __shared__ uint8_t fl[SR];
    const int chunk = blockIdx.x, tid = threadIdx.x;
    if (tid < SR) fl[tid] = start[chunk * SR + tid];
    __syncthreads();
    const int ch0 = tid * 4;
    const float* cp = carry + (size_t)chunk * H_ + ch0;
    float r0 = cp[0], r1 = cp[1], r2 = cp[2], r3 = cp[3];
    for (int r = 0; r < SR; ++r) {
        if (fl[r]) { r0 = r1 = r2 = r3 = 0.f; }
        const size_t base = (size_t)(chunk * SR + r) * H_ + ch0;
        v4bf v = *(const v4bf*)(in + base);
        r0 += (float)v[0]; r1 += (float)v[1]; r2 += (float)v[2]; r3 += (float)v[3];
        v4bf o; o[0] = f2bf(r0); o[1] = f2bf(r1); o[2] = f2bf(r2); o[3] = f2bf(r3);
        *(v4bf*)(z + base) = o;
    }
}

// ---------------- WMMA bf16 GEMM: C = act(A * W^T + bias) ----------------
// A: [M,K] (TA = float or bf16, row-major, lda). Optional second A source A2
// (same lda) taking over for k >= K1 (handles concat([z, emb]) without a copy).
// W: [N, ldw] bf16 row-major (K contiguous) -> exactly the B-fragment order.
template <typename TA, bool RELU, bool DUAL, typename TC>
__global__ __launch_bounds__(256) void k_gemm(
    const TA* __restrict__ A, const TA* __restrict__ A2, int lda,
    const bf16* __restrict__ W, int ldw, const float* __restrict__ bias,
    TC* __restrict__ Cout, int ldc, int M, int N, int K, int K1) {

    __shared__ __align__(16) bf16 As[BM][LDS_STRIDE];
    __shared__ __align__(16) bf16 Bs[BN][LDS_STRIDE];

    const int tid  = threadIdx.x;
    const int m0   = blockIdx.y * BM;
    const int n0   = blockIdx.x * BN;
    const int wv   = tid >> 5;
    const int lane = tid & 31;
    const int l15  = lane & 15;
    const int kh   = lane >> 4;            // which K-half this lane holds
    const int wm   = (wv & 3) * 32;        // wave sub-tile origin (4x2 wave grid)
    const int wn   = (wv >> 2) * 32;

    // tile-staging indices
    const int arow = tid >> 1, ahalf = tid & 1;  // A: 128 rows x 2 halves of 16 elems
    const int brow = tid >> 2, bseg  = tid & 3;  // B: 64 rows x 4 segs of 8 elems

    v8f acc[2][2];
#pragma unroll
    for (int i = 0; i < 2; ++i)
#pragma unroll
        for (int j = 0; j < 2; ++j)
#pragma unroll
            for (int e = 0; e < 8; ++e) acc[i][j][e] = 0.f;

    for (int k0 = 0; k0 < K; k0 += BK) {
        const TA* Ab; int kl;
        if (DUAL && k0 >= K1) { Ab = A2; kl = k0 - K1; } else { Ab = A; kl = k0; }

        // ---- stage A tile ----
        {
            const TA* src = Ab + (size_t)(m0 + arow) * lda + kl + ahalf * 16;
            if constexpr (sizeof(TA) == 4) {
                bf16 tmp[16];
                const v4f* p = (const v4f*)src;
#pragma unroll
                for (int q = 0; q < 4; ++q) {
                    v4f f = p[q];
                    tmp[q * 4 + 0] = f2bf(f[0]); tmp[q * 4 + 1] = f2bf(f[1]);
                    tmp[q * 4 + 2] = f2bf(f[2]); tmp[q * 4 + 3] = f2bf(f[3]);
                }
                *(v8bf*)&As[arow][ahalf * 16]     = *(v8bf*)&tmp[0];
                *(v8bf*)&As[arow][ahalf * 16 + 8] = *(v8bf*)&tmp[8];
            } else {
                *(v8bf*)&As[arow][ahalf * 16]     = *(const v8bf*)(src);
                *(v8bf*)&As[arow][ahalf * 16 + 8] = *(const v8bf*)(src + 8);
            }
        }
        // ---- stage B tile ----
        {
            const bf16* srcb = W + (size_t)(n0 + brow) * ldw + k0 + bseg * 8;
            *(v8bf*)&Bs[brow][bseg * 8] = *(const v8bf*)srcb;
        }
        // ---- prefetch next tiles (gfx1250 global_prefetch_b8) ----
        if (k0 + BK < K) {
            const TA* Abn; int kln;
            if (DUAL && (k0 + BK) >= K1) { Abn = A2; kln = k0 + BK - K1; }
            else                          { Abn = A;  kln = k0 + BK; }
            __builtin_prefetch(Abn + (size_t)(m0 + arow) * lda + kln + ahalf * 16, 0, 0);
            __builtin_prefetch(W   + (size_t)(n0 + brow) * ldw + k0 + BK + bseg * 8, 0, 0);
        }
        __syncthreads();

        // ---- WMMA: lane l15 row/col, K split by kh per ISA 16-bit layout ----
        v16bf af[2], bfg[2];
#pragma unroll
        for (int i = 0; i < 2; ++i) {
            const int r = wm + i * 16 + l15;
            af[i] = mk16(*(v8bf*)&As[r][8 * kh], *(v8bf*)&As[r][8 * kh + 16]);
        }
#pragma unroll
        for (int j = 0; j < 2; ++j) {
            const int r = wn + j * 16 + l15;
            bfg[j] = mk16(*(v8bf*)&Bs[r][8 * kh], *(v8bf*)&Bs[r][8 * kh + 16]);
        }
#pragma unroll
        for (int i = 0; i < 2; ++i)
#pragma unroll
            for (int j = 0; j < 2; ++j)
                acc[i][j] = __builtin_amdgcn_wmma_f32_16x16x32_bf16(
                    false, af[i], false, bfg[j], (short)0, acc[i][j], false, false);
        __syncthreads();
    }

    // ---- epilogue: C/D layout -> row = base + kh*8 + r, col = base + l15 ----
#pragma unroll
    for (int i = 0; i < 2; ++i)
#pragma unroll
        for (int j = 0; j < 2; ++j) {
            const int colg = n0 + wn + j * 16 + l15;
            const float bv = bias[colg];
#pragma unroll
            for (int r = 0; r < 8; ++r) {
                const int rowg = m0 + wm + i * 16 + kh * 8 + r;
                float v = acc[i][j][r] + bv;
                if (RELU) v = v > 0.f ? v : 0.01f * v;   // leaky_relu, slope 0.01
                storeC(&Cout[(size_t)rowg * ldc + colg], v);
            }
        }
}

// ---------------- host-side orchestration ----------------
extern "C" void kernel_launch(void* const* d_in, const int* in_sizes, int n_in,
                              void* d_out, int out_size, void* d_ws, size_t ws_size,
                              hipStream_t stream) {
    const float*   x        = (const float*)d_in[0];
    const uint8_t* start    = (const uint8_t*)d_in[1];
    const float*   h_states = (const float*)d_in[2];
    /* d_in[3] h_starts: flags on the prepended state never affect the sums */
    const float*   Win      = (const float*)d_in[4];
    const float*   b_in     = (const float*)d_in[5];
    const float*   proj_W   = (const float*)d_in[6];
    const float*   proj_b   = (const float*)d_in[7];
    const float*   ff_W     = (const float*)d_in[8];
    const float*   ff_b     = (const float*)d_in[9];
    const float*   Wout     = (const float*)d_in[10];
    const float*   b_out    = (const float*)d_in[11];
    float*         out      = (float*)d_out;

    char* ws = (char*)d_ws;
    size_t off = 0;
    auto alloc = [&](size_t bytes) -> void* {
        void* p = ws + off;
        off += (bytes + 255) & ~(size_t)255;
        return p;
    };

    bf16*  emb      = (bf16*)alloc((size_t)T_ * H_ * 2);
    bf16*  act      = (bf16*)alloc((size_t)T_ * H_ * 2);
    bf16*  zbuf     = (bf16*)alloc((size_t)T_ * H_ * 2);
    bf16*  zact     = (bf16*)alloc((size_t)T_ * H_ * 2);
    bf16*  Win_bf   = (bf16*)alloc((size_t)H_ * IN_ * 2);
    bf16*  projW_bf = (bf16*)alloc((size_t)L_ * H_ * H_ * 2);
    bf16*  ffW_bf   = (bf16*)alloc((size_t)L_ * H_ * 2 * H_ * 2);
    bf16*  Wout_bf  = (bf16*)alloc((size_t)OUT_ * H_ * 2);
    float* agg      = (float*)alloc((size_t)SC * H_ * 4);
    float* carry    = (float*)alloc((size_t)SC * H_ * 4);
    int*   aggflag  = (int*)alloc((size_t)SC * 4);

    // weights -> bf16 (once per call; cheap, ~26MB)
    k_f32_to_bf16<<<512,  256, 0, stream>>>(Win,    Win_bf,   H_ * IN_);
    k_f32_to_bf16<<<2048, 256, 0, stream>>>(proj_W, projW_bf, L_ * H_ * H_);
    k_f32_to_bf16<<<4096, 256, 0, stream>>>(ff_W,   ffW_bf,   L_ * H_ * 2 * H_);
    k_f32_to_bf16<<<512,  256, 0, stream>>>(Wout,   Wout_bf,  OUT_ * H_);

    const dim3 blk(256);
    // emb = x @ Win^T + b_in   (f32 A converted while staging to LDS)
    k_gemm<float, false, false, bf16><<<dim3(H_ / BN, T_ / BM), blk, 0, stream>>>(
        x, nullptr, IN_, Win_bf, IN_, b_in, emb, H_, T_, H_, IN_, 0);

    const bf16* layer_in = emb;
    for (int i = 0; i < L_; ++i) {
        // segmented scan over time
        k_scan_chunk_agg<<<SC, 256, 0, stream>>>(layer_in, start, agg, aggflag);
        k_scan_carries <<<1,  256, 0, stream>>>(agg, aggflag, h_states + (size_t)i * H_, carry);
        k_scan_apply   <<<SC, 256, 0, stream>>>(layer_in, start, carry, zbuf);

        // zact = lrelu(z @ proj_W^T + proj_b)
        k_gemm<bf16, true, false, bf16><<<dim3(H_ / BN, T_ / BM), blk, 0, stream>>>(
            zbuf, nullptr, H_, projW_bf + (size_t)i * H_ * H_, H_,
            proj_b + (size_t)i * H_, zact, H_, T_, H_, H_, 0);

        // act = lrelu(concat([zact, emb]) @ ff_W^T + ff_b)  (dual-A, K1 = H)
        k_gemm<bf16, true, true, bf16><<<dim3(H_ / BN, T_ / BM), blk, 0, stream>>>(
            zact, emb, H_, ffW_bf + (size_t)i * H_ * 2 * H_, 2 * H_,
            ff_b + (size_t)i * H_, act, H_, T_, H_, 2 * H_, H_);

        layer_in = act;
    }

    // out = act @ Wout^T + b_out  (f32 output)
    k_gemm<bf16, false, false, float><<<dim3(OUT_ / BN, T_ / BM), blk, 0, stream>>>(
        act, nullptr, H_, Wout_bf, H_, b_out, out, OUT_, T_, OUT_, H_, 0);
}
